// OptimizedTwoSidedVLUTAttention_29025388986486
// MI455X (gfx1250) — matrix operand order repaired and checked by reading, not compiled, not run on hardware
//
#include <hip/hip_runtime.h>
#include <hip/hip_bf16.h>
#include <math.h>

// ---------------------------------------------------------------------------
// MI455X (gfx1250) vLUT attention, round 3.
//  * QKV: single merged kernel (projection axis -> 6144 waves in one launch),
//    register-blocked 32x64 C-tile per wave, 8 v_wmma_f32_16x16x32_bf16/K-step,
//    global_prefetch on the A stream.
//  * Attention fused per (b,h,16-query stripe): scores (WMMA) -> LDS,
//    softmax in LDS (unnormalized exp; 1/rowsum folded into both stores),
//    attn_weights written to d_out once, PV reads A-frags from LDS.
//  * Out-proj: same blocked GEMM against sum_d(Wout) packed transposed bf16.
// ---------------------------------------------------------------------------

typedef __bf16 bf16_t;
typedef __attribute__((ext_vector_type(16))) __bf16 v16bf;
typedef __attribute__((ext_vector_type(8)))  float  v8f;
typedef __attribute__((ext_vector_type(4)))  float  v4f;

#define E_DIM 1024
#define D_DIM 8
#define NB    4
#define SEQ   1024
#define NH    16
#define HDIM  64
#define NROWS (NB * SEQ)        // 4096
#define KQKV  (E_DIM * D_DIM)   // 8192

#define SLICE_F (16 * SEQ + 32) // per-wave LDS slice (floats): scores + stats

__device__ __forceinline__ v8f wmma_bf16(v16bf a, v16bf b, v8f c) {
  return __builtin_amdgcn_wmma_f32_16x16x32_bf16(false, a, false, b,
                                                 (short)0, c, false, false);
}

// A-fragment (16x32, lane = row l&15, K-group l>>4): fp32 -> bf16.
// p already lane-adjusted; reads p[0..15].
__device__ __forceinline__ v16bf cvt_a_frag(const float* p) {
  v4f x0 = *(const v4f*)(p +  0);
  v4f x1 = *(const v4f*)(p +  4);
  v4f x2 = *(const v4f*)(p +  8);
  v4f x3 = *(const v4f*)(p + 12);
  v16bf a;
  #pragma unroll
  for (int i = 0; i < 4; ++i) {
    a[i]      = (bf16_t)x0[i];
    a[4 + i]  = (bf16_t)x1[i];
    a[8 + i]  = (bf16_t)x2[i];
    a[12 + i] = (bf16_t)x3[i];
  }
  return a;
}

// B-fragment (32x16, lane = col l&15, K-half l>>4): fp32 -> bf16,
// 16 contiguous values at p.
__device__ __forceinline__ v16bf cvt_b_frag(const float* p) {
  v4f x0 = *(const v4f*)(p + 0);
  v4f x1 = *(const v4f*)(p + 4);
  v4f x2 = *(const v4f*)(p + 8);
  v4f x3 = *(const v4f*)(p + 12);
  v16bf b;
  #pragma unroll
  for (int i = 0; i < 4; ++i) {
    b[i]      = (bf16_t)x0[i];
    b[4 + i]  = (bf16_t)x1[i];
    b[8 + i]  = (bf16_t)x2[i];
    b[12 + i] = (bf16_t)x3[i];
  }
  return b;
}

// Wave-level 32x64 tile body shared by the GEMM kernels.
__device__ __forceinline__ void gemm_tile_32x64(
    const float* __restrict__ A, int lda,
    const bf16_t* __restrict__ Bt, int ldb,
    const float* __restrict__ bias,
    float* __restrict__ C, int ldc,
    int tm, int tn, int K, int lane) {
  const int cl = lane & 15, kh = lane >> 4;

  const float*  arow0 = A + (size_t)(tm * 32 + cl) * lda + kh * 16;
  const float*  arow1 = arow0 + (size_t)16 * lda;
  const bf16_t* brow  = Bt + (size_t)(tn * 64 + cl) * ldb + kh * 16;

  v8f acc[2][4] = {};
  for (int k0 = 0; k0 < K; k0 += 32) {
    __builtin_prefetch(arow0 + k0 + 512, 0, 1);   // global_prefetch, ~16 steps ahead
    __builtin_prefetch(arow1 + k0 + 512, 0, 1);
    v16bf a0 = cvt_a_frag(arow0 + k0);
    v16bf a1 = cvt_a_frag(arow1 + k0);
    #pragma unroll
    for (int j = 0; j < 4; ++j) {
      v16bf b = *(const v16bf*)(brow + (size_t)(j * 16) * ldb + k0);
      acc[0][j] = wmma_bf16(a0, b, acc[0][j]);
      acc[1][j] = wmma_bf16(a1, b, acc[1][j]);
    }
  }

  #pragma unroll
  for (int i = 0; i < 2; ++i) {
    #pragma unroll
    for (int j = 0; j < 4; ++j) {
      const int   fcol = tn * 64 + j * 16 + cl;
      const float badd = bias[fcol];
      float* crow = C + (size_t)(tm * 32 + i * 16 + 8 * kh) * ldc + fcol;
      #pragma unroll
      for (int r = 0; r < 8; ++r) crow[(size_t)r * ldc] = acc[i][j][r] + badd;
    }
  }
}

// ------------------- merged QKV projection (one launch, 6144 waves) ---------
// q/k/v[4096,1024] = X[4096,8192] @ Wt{q,k,v}[1024,8192]^T + bias.
__global__ __launch_bounds__(256)
void gemm_qkv(const float* __restrict__ A,
              const bf16_t* __restrict__ WtQ, const bf16_t* __restrict__ WtK,
              const bf16_t* __restrict__ WtV,
              const float* __restrict__ bQ, const float* __restrict__ bK,
              const float* __restrict__ bV,
              float* __restrict__ Cq, float* __restrict__ Ck,
              float* __restrict__ Cv) {
  const int lane = threadIdx.x & 31;
  const int wave = (blockIdx.x * blockDim.x + threadIdx.x) >> 5;
  const int proj = wave >> 11;              // 0:q 1:k 2:v (wave-uniform)
  const int t    = wave & 2047;
  const int tm   = t >> 4;                  // 128 row-blocks of 32
  const int tn   = t & 15;                  // 16 col-blocks of 64

  const bf16_t* Bt   = (proj == 0) ? WtQ : (proj == 1) ? WtK : WtV;
  const float*  bias = (proj == 0) ? bQ  : (proj == 1) ? bK  : bV;
  float*        C    = (proj == 0) ? Cq  : (proj == 1) ? Ck  : Cv;

  gemm_tile_32x64(A, KQKV, Bt, KQKV, bias, C, E_DIM, tm, tn, KQKV, lane);
}

// ------------------- generic blocked GEMM (out-projection) ------------------
__global__ __launch_bounds__(256)
void gemm_a32_b16(const float* __restrict__ A, int lda,
                  const bf16_t* __restrict__ Bt, int ldb,
                  const float* __restrict__ bias,
                  float* __restrict__ C, int ldc,
                  int tiles_n, int K) {
  const int lane = threadIdx.x & 31;
  const int wave = (blockIdx.x * blockDim.x + threadIdx.x) >> 5;
  gemm_tile_32x64(A, lda, Bt, ldb, bias, C, ldc,
                  wave / tiles_n, wave % tiles_n, K, lane);
}

// ------------------- fused attention: scores -> softmax -> PV ---------------
// One wave handles one (b,h) and one 16-query stripe.
// q,k: [4096,1024] fp32; Vt: [64][64][1024] bf16; attnw: [64][1024][1024];
// O: [4096,1024] fp32 (merged heads).
__global__ __launch_bounds__(128)
void attn_fused(const float* __restrict__ q, const float* __restrict__ k,
                const bf16_t* __restrict__ Vt,
                float* __restrict__ attnw, float* __restrict__ O,
                float scale) {
  extern __shared__ float lds_f[];
  const int lane = threadIdx.x & 31;
  const int wib  = threadIdx.x >> 5;                       // wave in block (0..3)
  const int wave = blockIdx.x * 4 + wib;
  const int bh   = wave >> 6;                              // 64 (b,h) pairs
  const int strp = wave & 63;                              // 16-query stripe
  const int b = bh >> 4, h = bh & 15;
  const int qs0 = strp * 16;
  const int cl = lane & 15, kh = lane >> 4;

  float* S        = lds_f + wib * SLICE_F;                 // [16][1024] scores
  float* row_stat = S + 16 * SEQ;                          // [16] 1/rowsum

  // ---- phase 1: scores tile row -> LDS (scaled) ----
  const float* aq = q + (size_t)(b * SEQ + qs0 + cl) * E_DIM + h * HDIM + kh * 16;
  const v16bf aq0 = cvt_a_frag(aq);        // K = 0..31 of head dim
  const v16bf aq1 = cvt_a_frag(aq + 32);   // K = 32..63
  for (int kt = 0; kt < 64; ++kt) {
    const float* kb = k + (size_t)(b * SEQ + kt * 16 + cl) * E_DIM + h * HDIM + kh * 16;
    v8f acc = {};
    acc = wmma_bf16(aq0, cvt_b_frag(kb),      acc);
    acc = wmma_bf16(aq1, cvt_b_frag(kb + 32), acc);
    float* srow = S + (size_t)(8 * kh) * SEQ + kt * 16 + cl;
    #pragma unroll
    for (int r = 0; r < 8; ++r) srow[(size_t)r * SEQ] = acc[r] * scale;
  }
  __syncthreads();

  // ---- phase 2: softmax rows in LDS; keep unnormalized exp, record 1/sum ---
  for (int r = 0; r < 16; ++r) {
    float* row = S + (size_t)r * SEQ;
    float m = -3.402823e38f;
    #pragma unroll
    for (int j = 0; j < 8; ++j) {
      v4f vv = *(const v4f*)(row + lane * 4 + j * 128);
      #pragma unroll
      for (int i = 0; i < 4; ++i) m = fmaxf(m, vv[i]);
    }
    #pragma unroll
    for (int off = 16; off > 0; off >>= 1) m = fmaxf(m, __shfl_xor(m, off, 32));
    float s = 0.0f;
    #pragma unroll
    for (int j = 0; j < 8; ++j) {
      v4f vv = *(const v4f*)(row + lane * 4 + j * 128);
      v4f ev;
      #pragma unroll
      for (int i = 0; i < 4; ++i) { ev[i] = __expf(vv[i] - m); s += ev[i]; }
      *(v4f*)(row + lane * 4 + j * 128) = ev;
    }
    #pragma unroll
    for (int off = 16; off > 0; off >>= 1) s += __shfl_xor(s, off, 32);
    if (lane == 0) row_stat[r] = 1.0f / s;
  }
  __syncthreads();

  // ---- phase 2b: write normalized attention weights to d_out (once) -------
  float* gdst = attnw + ((size_t)bh << 20) + (size_t)qs0 * SEQ;
  for (int r = 0; r < 16; ++r) {
    const float inv = row_stat[r];
    #pragma unroll
    for (int j = 0; j < 8; ++j) {
      v4f vv = *(const v4f*)(S + (size_t)r * SEQ + lane * 4 + j * 128);
      v4f ov;
      #pragma unroll
      for (int i = 0; i < 4; ++i) ov[i] = vv[i] * inv;
      *(v4f*)(gdst + (size_t)r * SEQ + lane * 4 + j * 128) = ov;
    }
  }
  __syncthreads();

  // ---- phase 3: O(16x64) = P(16x1024, LDS) @ V(1024x64); fold 1/sum -------
  const float* alds = S + (size_t)cl * SEQ + kh * 16;
  const bf16_t* bvt = Vt + ((size_t)bh << 16) + (size_t)cl * SEQ + kh * 16;
  v8f acc[4] = {};
  for (int k0 = 0; k0 < SEQ; k0 += 32) {
    v16bf af = cvt_a_frag(alds + k0);          // ds_load_b128 + cvt
    #pragma unroll
    for (int j = 0; j < 4; ++j) {
      v16bf bf_ = *(const v16bf*)(bvt + (size_t)(j * 16) * SEQ + k0);
      acc[j] = wmma_bf16(af, bf_, acc[j]);
    }
  }
  #pragma unroll
  for (int r = 0; r < 8; ++r) {
    const int   m   = qs0 + r + 8 * kh;
    const float inv = row_stat[r + 8 * kh];
    float* orow = O + (size_t)(b * SEQ + m) * E_DIM + h * HDIM;
    #pragma unroll
    for (int j = 0; j < 4; ++j) orow[j * 16 + cl] = acc[j][r] * inv;
  }
}

// ------------------------- packing kernels ----------------------------------
// W[e][f][d] fp32 -> Wt[f][e*8+d] bf16 (transposed, K-contiguous B frags)
__global__ __launch_bounds__(256)
void pack_w_qkv(const float* __restrict__ W, bf16_t* __restrict__ Wt) {
  const int id = blockIdx.x * blockDim.x + threadIdx.x;   // 1<<20
  const int f = id & (E_DIM - 1);
  const int e = id >> 10;
  const float* src = W + ((size_t)e * E_DIM + f) * D_DIM;
  bf16_t* dst = Wt + ((size_t)f * E_DIM + e) * D_DIM;
  #pragma unroll
  for (int d = 0; d < D_DIM; ++d) dst[d] = (bf16_t)src[d];
}

// Wout[e][f][d] fp32 -> Wsum^T[f][e] bf16 (sum over d)
__global__ __launch_bounds__(256)
void pack_wsum(const float* __restrict__ W, bf16_t* __restrict__ Wt2) {
  const int id = blockIdx.x * blockDim.x + threadIdx.x;   // 1<<20
  const int f = id & (E_DIM - 1);
  const int e = id >> 10;
  const float* src = W + ((size_t)e * E_DIM + f) * D_DIM;
  float s = 0.0f;
  #pragma unroll
  for (int d = 0; d < D_DIM; ++d) s += src[d];
  Wt2[(size_t)f * E_DIM + e] = (bf16_t)s;
}

// v[4096][1024] fp32 -> Vt[b][h][hd][s] bf16
__global__ __launch_bounds__(256)
void pack_vt(const float* __restrict__ V, bf16_t* __restrict__ Vt) {
  const int id = blockIdx.x * blockDim.x + threadIdx.x;   // 1<<22
  const int n = id >> 10;
  const int e = id & (E_DIM - 1);
  const int b = n >> 10, s = n & (SEQ - 1);
  const int h = e >> 6,  hd = e & (HDIM - 1);
  Vt[((size_t)((b * NH + h) * HDIM + hd) << 10) + s] = (bf16_t)V[(size_t)n * E_DIM + e];
}

// ---------------------------------------------------------------------------
extern "C" void kernel_launch(void* const* d_in, const int* in_sizes, int n_in,
                              void* d_out, int out_size, void* d_ws, size_t ws_size,
                              hipStream_t stream) {
  (void)in_sizes; (void)n_in; (void)out_size; (void)ws_size;
  const float* x  = (const float*)d_in[0];
  const float* wq = (const float*)d_in[1];
  const float* wk = (const float*)d_in[2];
  const float* wv = (const float*)d_in[3];
  const float* wo = (const float*)d_in[4];
  const float* bq = (const float*)d_in[5];
  const float* bk = (const float*)d_in[6];
  const float* bv = (const float*)d_in[7];
  const float* bo = (const float*)d_in[8];

  char* ws = (char*)d_ws;
  size_t off = 0;
  auto take = [&](size_t bytes) -> void* {
    void* p = ws + off;
    off = (off + bytes + 255) & ~(size_t)255;
    return p;
  };
  bf16_t* wtq  = (bf16_t*)take((size_t)KQKV * E_DIM * sizeof(bf16_t));   // 16.8 MB
  bf16_t* wtk  = (bf16_t*)take((size_t)KQKV * E_DIM * sizeof(bf16_t));
  bf16_t* wtv  = (bf16_t*)take((size_t)KQKV * E_DIM * sizeof(bf16_t));
  bf16_t* wt2  = (bf16_t*)take((size_t)E_DIM * E_DIM * sizeof(bf16_t));  // 2 MB
  float*  qf   = (float*)take((size_t)NROWS * E_DIM * sizeof(float));    // 16.8 MB
  float*  kf   = (float*)take((size_t)NROWS * E_DIM * sizeof(float));
  float*  vf   = (float*)take((size_t)NROWS * E_DIM * sizeof(float));
  bf16_t* vt   = (bf16_t*)take((size_t)NB * NH * HDIM * SEQ * sizeof(bf16_t)); // 8.4 MB
  float*  aout = (float*)take((size_t)NROWS * E_DIM * sizeof(float));    // 16.8 MB

  float* out0 = (float*)d_out;                        // [4096][1024]
  float* attn = out0 + (size_t)NROWS * E_DIM;         // [64][1024][1024]

  const dim3 blk(256);

  // 1) pack weights to transposed bf16
  pack_w_qkv<<<4096, blk, 0, stream>>>(wq, wtq);
  pack_w_qkv<<<4096, blk, 0, stream>>>(wk, wtk);
  pack_w_qkv<<<4096, blk, 0, stream>>>(wv, wtv);
  pack_wsum <<<4096, blk, 0, stream>>>(wo, wt2);

  // 2) merged QKV projections: 3 x (128x16) tiles of 32x64 = 6144 waves
  //    in one launch -> 768 blocks of 8 waves.
  gemm_qkv<<<768, blk, 0, stream>>>(x, wtq, wtk, wtv, bq, bk, bv, qf, kf, vf);

  // 3) fused attention: 64 (b,h) x 64 stripes = 4096 waves, 4 waves/block,
  //    dynamic LDS = 4 slices of (16*1024+32) floats (~257 KB / WGP).
  pack_vt<<<16384, blk, 0, stream>>>(vf, vt);
  const size_t lds_bytes = 4 * (size_t)SLICE_F * sizeof(float);
  attn_fused<<<1024, dim3(128), lds_bytes, stream>>>(qf, kf, vt, attn, aout, 0.125f);

  // 4) output projection: [4096,1024]@[1024,1024]+bias; 128x16 tiles of 32x64
  gemm_a32_b16<<<256, blk, 0, stream>>>(aout, E_DIM, wt2, E_DIM, bo, out0, E_DIM, 16, E_DIM);
}
